// DCNv4_pytorch_46127948759509
// MI455X (gfx1250) — matrix operand order repaired and checked by reading, not compiled, not run on hardware
//
#include <hip/hip_runtime.h>

typedef __attribute__((ext_vector_type(16))) _Float16 v16h;
typedef __attribute__((ext_vector_type(8)))  _Float16 v8h;
typedef __attribute__((ext_vector_type(4)))  _Float16 v4h;
typedef __attribute__((ext_vector_type(2)))  _Float16 v2h;
typedef __attribute__((ext_vector_type(8)))  float    v8f;

#define BM 128
#define BN 64
#define BK 64

// ---------------- pack weights: concat + transpose + f32->f16 (one shot, tiny) ----------------
// Wcat_t: (384 x 256) f16, row n, col k; n in [0,256)=w_value, [256,328)=w_offset,
// [328,364)=w_mask, [364,384)=zero pad. Wout_t: (256 x 256) f16 transposed. bcat: 384 f32.
__global__ __launch_bounds__(256) void pack_weights(
    const float* __restrict__ wv, const float* __restrict__ wo,
    const float* __restrict__ wm, const float* __restrict__ wout,
    const float* __restrict__ bv, const float* __restrict__ bo,
    const float* __restrict__ bm,
    _Float16* __restrict__ Wcat_t, _Float16* __restrict__ Wout_t,
    float* __restrict__ bcat)
{
    int idx = blockIdx.x * 256 + threadIdx.x;
    if (idx < 384 * 256) {
        int nn = idx >> 8, k = idx & 255;
        float v = 0.0f;
        if (nn < 256)      v = wv[k * 256 + nn];
        else if (nn < 328) v = wo[k * 72 + (nn - 256)];
        else if (nn < 364) v = wm[k * 36 + (nn - 328)];
        Wcat_t[idx] = (_Float16)v;
        if (k == 0) {
            float b = 0.0f;
            if (nn < 256)      b = bv[nn];
            else if (nn < 328) b = bo[nn - 256];
            else if (nn < 364) b = bm[nn - 328];
            bcat[nn] = b;
        }
    } else {
        int j = idx - 384 * 256;
        if (j < 256 * 256) {
            int nn = j >> 8, k = j & 255;
            Wout_t[j] = (_Float16)wout[k * 256 + nn];
        }
    }
}

// ---------------- x: f32 -> f16, vectorized ----------------
__global__ __launch_bounds__(256) void cvt_f32_to_f16(
    const float* __restrict__ in, _Float16* __restrict__ out, int n4)
{
    int i = blockIdx.x * 256 + threadIdx.x;
    if (i < n4) {
        float4 v = ((const float4*)in)[i];
        v4h h = { (_Float16)v.x, (_Float16)v.y, (_Float16)v.z, (_Float16)v.w };
        ((v4h*)out)[i] = h;
    }
}

// ---------------- WMMA GEMM: C[M,NTOT] = A[M,256](f16) * Bt[NTOT,256](f16,transposed) + bias ----
// Compile-time NTOT/K: all strides constant, all tiles full, no bounds checks.
// 8 wave32s; wave grid 4(M) x 2(N); each wave = 32x32 output = 2x2 WMMA fragments.
// BK=64 => 8 v_wmma per barrier pair.
template<int NTOT>
__global__ __launch_bounds__(256) void gemm_h16_wmma(
    const _Float16* __restrict__ A,
    const _Float16* __restrict__ Bt,
    const float* __restrict__ bias,
    float* __restrict__ C)
{
    constexpr int K = 256;
    __shared__ __align__(32) _Float16 sA[BM][BK];   // 16 KB
    __shared__ __align__(32) _Float16 sB[BN][BK];   // 8 KB

    constexpr int nBlocksN = NTOT / BN;
    const int m0 = (blockIdx.x / nBlocksN) * BM;
    const int n0 = (blockIdx.x % nBlocksN) * BN;

    const int tid  = threadIdx.x;
    const int lane = tid & 31;
    const int wave = tid >> 5;
    const int wm = (wave >> 1) * 32;
    const int wn = (wave & 1) * 32;

    // staging geometry: chunk = 8 halfs (16B); 8 chunks per 64-half row.
    const int sr = tid >> 3;            // 0..31
    const int sc = (tid & 7) * 8;       // 0..56
    const _Float16* pa0 = A  + (size_t)(m0 + sr) * K + sc;
    const _Float16* pb0 = Bt + (size_t)(n0 + sr) * K + sc;

    // fragment geometry (CDNA5 wave32 WMMA VGPR layouts)
    const int arow = lane & 15;
    const int akb  = (lane >> 4) * 8;
    const int bcol = lane & 15;
    const int bkb  = (lane >> 4) * 8;

    v8f acc[2][2] = {};

    for (int k0 = 0; k0 < K; k0 += BK) {
        #pragma unroll
        for (int t = 0; t < 4; ++t)     // A: 128 rows in 4 slabs of 32
            *(v8h*)&sA[sr + 32 * t][sc] = *(const v8h*)(pa0 + (size_t)(32 * t) * K);
        #pragma unroll
        for (int t = 0; t < 2; ++t)     // B: 64 rows in 2 slabs of 32
            *(v8h*)&sB[sr + 32 * t][sc] = *(const v8h*)(pb0 + (size_t)(32 * t) * K);
        __builtin_prefetch(pa0 + BK, 0, 3);   // next k-tile -> global_prefetch_b8
        __builtin_prefetch(pb0 + BK, 0, 3);
        pa0 += BK; pb0 += BK;
        __syncthreads();

        #pragma unroll
        for (int s = 0; s < 2; ++s) {          // two 16x16x32 k-steps per LDS tile
            v16h afrag[2], bfrag[2];
            #pragma unroll
            for (int i = 0; i < 2; ++i) {
                const _Float16* pa = &sA[wm + i * 16 + arow][s * 32];
                v8h lo = *(const v8h*)(pa + akb);        // K akb..+7     -> VGPRs 0-3
                v8h hi = *(const v8h*)(pa + akb + 16);   // K akb+16..+23 -> VGPRs 4-7
                #pragma unroll
                for (int t = 0; t < 8; ++t) { afrag[i][t] = lo[t]; afrag[i][t + 8] = hi[t]; }
            }
            #pragma unroll
            for (int j = 0; j < 2; ++j) {
                const _Float16* pbf = &sB[wn + j * 16 + bcol][s * 32];
                v8h lo = *(const v8h*)(pbf + bkb);
                v8h hi = *(const v8h*)(pbf + bkb + 16);
                #pragma unroll
                for (int t = 0; t < 8; ++t) { bfrag[j][t] = lo[t]; bfrag[j][t + 8] = hi[t]; }
            }
            #pragma unroll
            for (int i = 0; i < 2; ++i)
                #pragma unroll
                for (int j = 0; j < 2; ++j)
                    acc[i][j] = __builtin_amdgcn_wmma_f32_16x16x32_f16(
                        false, afrag[i], false, bfrag[j],
                        (short)0, acc[i][j], false, false);
        }
        __syncthreads();
    }

    // C/D layout: VGPR r -> row (lane>>4)*8 + r, col = lane & 15. NTOT constant => imm offsets.
    const int crow = (lane >> 4) * 8;
    const int ccol = lane & 15;
    #pragma unroll
    for (int i = 0; i < 2; ++i) {
        #pragma unroll
        for (int j = 0; j < 2; ++j) {
            int gn = n0 + wn + j * 16 + ccol;
            float bvl = bias[gn];
            float* cp = C + (size_t)(m0 + wm + i * 16 + crow) * NTOT + gn;
            #pragma unroll
            for (int r = 0; r < 8; ++r) {
                cp[r * NTOT] = acc[i][j][r] + bvl;
            }
        }
    }
}

// ---------------- DCNv4 core: fused vom(384-wide) -> y(f16) ----------------
// vom row layout: [0,256)=value, [256,328)=offset(G,P,2), [328,364)=mask(G,P).
// gx = w + iw + off_x, gy = h + ih + off_y in padded coords; zero pad ring =>
// exact bounds check against the unpadded 64x64 image.
__global__ __launch_bounds__(256) void dcnv4_core(
    const float* __restrict__ vom,
    _Float16* __restrict__ yh)       // (N,4096,256) f16
{
    const int wid  = blockIdx.x * 8 + (threadIdx.x >> 5);
    const int lane = threadIdx.x & 31;
    const int g = wid & 3;
    const int q = (wid >> 2) & 4095;
    const int n = wid >> 14;
    const int h = q >> 6, w = q & 63;

    const size_t pix = (size_t)n * 4096 + q;
    const float* offp  = vom + pix * 384 + 256 + g * 18;
    const float* mskp  = vom + pix * 384 + 328 + g * 9;
    const float* vbase = vom + (size_t)n * 4096 * 384 + g * 64 + lane * 2;

    float acc0 = 0.0f, acc1 = 0.0f;

    #pragma unroll
    for (int p = 0; p < 9; ++p) {
        const int iw = p / 3, ih = p % 3;                // iw-major flattening
        const float gx = (float)(w + iw) + offp[2 * p];
        const float gy = (float)(h + ih) + offp[2 * p + 1];
        const float m  = mskp[p];

        const float x0f = floorf(gx), y0f = floorf(gy);
        const float wx = gx - x0f, wy = gy - y0f;
        const int x0 = (int)x0f - 1;                     // unpadded coords
        const int y0 = (int)y0f - 1;

        const float cw[4] = { (1.f - wx) * (1.f - wy), wx * (1.f - wy),
                              (1.f - wx) * wy,          wx * wy };
        const int xs[4] = { x0, x0 + 1, x0,     x0 + 1 };
        const int ys[4] = { y0, y0,     y0 + 1, y0 + 1 };

        #pragma unroll
        for (int c = 0; c < 4; ++c) {
            const int xi = xs[c], yi = ys[c];
            if (xi >= 0 && xi < 64 && yi >= 0 && yi < 64) {
                const float2 v2 = *(const float2*)(vbase + (size_t)(yi * 64 + xi) * 384);
                const float wgt = cw[c] * m;
                acc0 += wgt * v2.x;
                acc1 += wgt * v2.y;
            }
        }
    }

    v2h st = { (_Float16)acc0, (_Float16)acc1 };
    *(v2h*)(yh + pix * 256 + g * 64 + lane * 2) = st;
}

// ---------------- launch ----------------
extern "C" void kernel_launch(void* const* d_in, const int* in_sizes, int n_in,
                              void* d_out, int out_size, void* d_ws, size_t ws_size,
                              hipStream_t stream) {
    const float* x        = (const float*)d_in[0];
    const float* w_value  = (const float*)d_in[1];
    const float* b_value  = (const float*)d_in[2];
    const float* w_offset = (const float*)d_in[3];
    const float* b_offset = (const float*)d_in[4];
    const float* w_mask   = (const float*)d_in[5];
    const float* b_mask   = (const float*)d_in[6];
    const float* w_out    = (const float*)d_in[7];
    const float* b_out    = (const float*)d_in[8];
    float* out = (float*)d_out;

    const int M = 8 * 64 * 64;   // 32768
    const int K = 256;

    char* ws = (char*)d_ws;
    _Float16* xh    = (_Float16*)(ws);                                   // 16 MB
    float*    vom   = (float*)   (ws + (size_t)16777216);                // 50.3 MB
    _Float16* yh    = (_Float16*)(ws + (size_t)16777216 + 50331648);     // 16 MB
    _Float16* Wcat  = (_Float16*)(ws + (size_t)83886080);                // 384x256 f16
    _Float16* Woutt = Wcat + 384 * 256;                                  // 256x256 f16
    float*    bcat  = (float*)(Woutt + 256 * 256);                       // 384 f32

    dim3 blk(256);

    pack_weights<<<(384 * 256 + 256 * 256) / 256, blk, 0, stream>>>(
        w_value, w_offset, w_mask, w_out, b_value, b_offset, b_mask,
        Wcat, Woutt, bcat);

    cvt_f32_to_f16<<<(M * K / 4) / 256, blk, 0, stream>>>(x, xh, M * K / 4);

    // fused value|offset|mask projection: (32768 x 256) x (256 x 384)
    gemm_h16_wmma<384><<<(M / BM) * (384 / BN), blk, 0, stream>>>(xh, Wcat, bcat, vom);

    // deformable sampling: one wave per (n, pixel, group)
    dcnv4_core<<<(8 * 4096 * 4) / 8, blk, 0, stream>>>(vom, yh);

    // output projection: (32768 x 256) x (256 x 256)
    gemm_h16_wmma<256><<<(M / BM) * (256 / BN), blk, 0, stream>>>(yh, Woutt, b_out, out);
}
